// QLinearLR_38379827757513
// MI455X (gfx1250) — matrix operand order repaired and checked by reading, not compiled, not run on hardware
//
#include <hip/hip_runtime.h>
#include <stdint.h>

#define IN_F   4096
#define OUT_F  4096
#define RANK   64
#define M_TOT  8192
#define STRIDE 44   // dwords per LDS row (88 bf16): 16B-alignable + conflict-free

typedef __attribute__((ext_vector_type(16))) __bf16 v16bf;
typedef __attribute__((ext_vector_type(2)))  __bf16 v2bf;
typedef __attribute__((ext_vector_type(2)))  float  v2f;
typedef __attribute__((ext_vector_type(8)))  float  v8f;

__device__ __constant__ float NF4T[16] = {
  -1.0f, -0.6961928009986877f, -0.5250730514526367f, -0.39491748809814453f,
  -0.28444138169288635f, -0.18477343022823334f, -0.09105003625154495f, 0.0f,
  0.07958029955625534f, 0.16093020141124725f, 0.24611230194568634f,
  0.33791524171829224f, 0.44070982933044434f, 0.5626170039176941f,
  0.7229568362236023f, 1.0f };

union Frag { v16bf v; uint32_t u[8]; uint4 q[2]; };

// two f32 -> packed bf16 pair (expect v_cvt_pk_bf16_f32)
__device__ __forceinline__ uint32_t cvt2(float a, float b) {
  v2f f = {a, b};
  v2bf h = __builtin_convertvector(f, v2bf);
  return __builtin_bit_cast(uint32_t, h);
}
// packed bf16 pair * packed bf16 scale (expect v_pk_mul_bf16)
__device__ __forceinline__ uint32_t pkmul(uint32_t t, v2bf sv) {
  v2bf tv = __builtin_bit_cast(v2bf, t);
  v2bf r  = tv * sv;
  return __builtin_bit_cast(uint32_t, r);
}
__device__ __forceinline__ v8f vzero() {
  v8f z = {0.f,0.f,0.f,0.f,0.f,0.f,0.f,0.f};
  return z;
}

// ---------------------------------------------------------------------------
// Kernel 1: T[8192x64] = x[8192x4096] @ lora_A[64x4096]^T   (bf16 WMMA)
// block: 256 thr = 8 waves in 4(M)x2(N); wave tile 32x32 -> 2x2 frags
// ---------------------------------------------------------------------------
__global__ __launch_bounds__(256) void lora_ax_kernel(
    const float* __restrict__ x, const float* __restrict__ lora_A,
    float* __restrict__ T)
{
  __shared__ __align__(16) uint32_t Xs[128 * STRIDE];
  __shared__ __align__(16) uint32_t Ls[64 * STRIDE];

  const int tid  = threadIdx.x;
  const int lane = tid & 31;
  const int wv   = tid >> 5;
  const int wm   = wv >> 1;            // 0..3
  const int wn   = wv & 1;             // 0..1
  const int hf   = lane >> 4;
  const int ln   = lane & 15;
  const int bm   = blockIdx.x;

  const int row  = tid >> 1;           // 0..127
  const int rhf  = tid & 1;
  const int lrow = tid >> 2;           // 0..63
  const int lq   = tid & 3;

  const size_t xbase = (size_t)(bm * 128 + row) * IN_F + rhf * 32;
  const size_t lbase = (size_t)lrow * IN_F + lq * 16;

  v8f acc[2][2];
  for (int i = 0; i < 2; ++i) for (int j = 0; j < 2; ++j) acc[i][j] = vzero();

  for (int kt = 0; kt < IN_F; kt += 64) {
    __syncthreads();
    { // stage x tile 128x64 f32 -> bf16
      const float4* xr = (const float4*)(x + xbase + kt);
      uint32_t* dst = &Xs[row * STRIDE + rhf * 16];
      #pragma unroll
      for (int j = 0; j < 4; ++j) {
        float4 f0 = xr[2*j], f1 = xr[2*j+1];
        uint4 o = make_uint4(cvt2(f0.x,f0.y), cvt2(f0.z,f0.w),
                             cvt2(f1.x,f1.y), cvt2(f1.z,f1.w));
        *(uint4*)(dst + 4*j) = o;
      }
    }
    { // stage lora_A tile 64x64 f32 -> bf16
      const float4* lr = (const float4*)(lora_A + lbase + kt);
      uint32_t* dst = &Ls[lrow * STRIDE + lq * 8];
      #pragma unroll
      for (int j = 0; j < 2; ++j) {
        float4 f0 = lr[2*j], f1 = lr[2*j+1];
        uint4 o = make_uint4(cvt2(f0.x,f0.y), cvt2(f0.z,f0.w),
                             cvt2(f1.x,f1.y), cvt2(f1.z,f1.w));
        *(uint4*)(dst + 4*j) = o;
      }
    }
    if (kt + 64 < IN_F) __builtin_prefetch(x + xbase + kt + 64, 0, 1);
    __syncthreads();

    #pragma unroll
    for (int kk = 0; kk < 2; ++kk) {
      const int kb = kk * 16;
      Frag a[2], b[2];
      #pragma unroll
      for (int fm = 0; fm < 2; ++fm) {
        const uint32_t* p = &Xs[(wm*32 + fm*16 + ln) * STRIDE + kb + hf*4];
        a[fm].q[0] = *(const uint4*)p;
        a[fm].q[1] = *(const uint4*)(p + 8);
      }
      #pragma unroll
      for (int fn = 0; fn < 2; ++fn) {
        const uint32_t* p = &Ls[(wn*32 + fn*16 + ln) * STRIDE + kb + hf*4];
        b[fn].q[0] = *(const uint4*)p;
        b[fn].q[1] = *(const uint4*)(p + 8);
      }
      #pragma unroll
      for (int fm = 0; fm < 2; ++fm)
        #pragma unroll
        for (int fn = 0; fn < 2; ++fn)
          acc[fm][fn] = __builtin_amdgcn_wmma_f32_16x16x32_bf16(
              false, a[fm].v, false, b[fn].v, (short)0, acc[fm][fn], false, false);
    }
  }

  #pragma unroll
  for (int fn = 0; fn < 2; ++fn) {
    const int rg = wn*32 + fn*16 + ln;
    #pragma unroll
    for (int fm = 0; fm < 2; ++fm)
      #pragma unroll
      for (int v = 0; v < 8; ++v) {
        const int mg = bm*128 + wm*32 + fm*16 + hf*8 + v;
        T[(size_t)mg * RANK + rg] = acc[fm][fn][v];
      }
  }
}

// ---------------------------------------------------------------------------
// Kernel 2: out = x @ dequant(qweight)^T  + T @ lora_B^T   (fused epilogue)
// block: 256 thr = 8 waves in 2(M)x4(N); wave tile 64x32 -> 4x2 frags
// NF4 dequant via 256-entry LDS LUT of pre-packed bf16 pairs + v_pk_mul_bf16
// ---------------------------------------------------------------------------
__global__ __launch_bounds__(256) void qgemm_lora_kernel(
    const float* __restrict__ x, const int* __restrict__ qweight,
    const float* __restrict__ absmax, const float* __restrict__ Tlora,
    const float* __restrict__ lora_B, float* __restrict__ out)
{
  __shared__ __align__(16) uint32_t As[128 * STRIDE];
  __shared__ __align__(16) uint32_t Bs[128 * STRIDE];
  __shared__ uint32_t LUT[256];        // byte -> packed {bf16(lo nib), bf16(hi nib)}

  const int tid  = threadIdx.x;
  const int lane = tid & 31;
  const int wv   = tid >> 5;
  const int wm   = wv >> 2;            // 0..1
  const int wn   = wv & 3;             // 0..3
  const int hf   = lane >> 4;
  const int ln   = lane & 15;

  const int bn = blockIdx.x;           // 0..31 (N tiles) — inner for x reuse in L2
  const int bm = blockIdx.y;           // 0..63 (M tiles)

  const int row = tid >> 1;            // 0..127
  const int rhf = tid & 1;

  // one-time LUT build (256 threads == 256 entries); covered by first barrier
  LUT[tid] = cvt2(NF4T[tid & 15], NF4T[(tid >> 4) & 15]);

  const size_t xbase = (size_t)(bm * 128 + row) * IN_F + rhf * 32;
  const size_t qbase = (size_t)(bn * 128 + row) * (IN_F / 2) + rhf * 16;
  const int    abase = (bn * 128 + row) * (IN_F / 64);

  v8f acc[4][2];
  for (int i = 0; i < 4; ++i) for (int j = 0; j < 2; ++j) acc[i][j] = vzero();

  auto mma_steps = [&]() {
    #pragma unroll
    for (int kk = 0; kk < 2; ++kk) {
      const int kb = kk * 16;
      Frag a[4], b[2];
      #pragma unroll
      for (int fm = 0; fm < 4; ++fm) {
        const uint32_t* p = &As[(wm*64 + fm*16 + ln) * STRIDE + kb + hf*4];
        a[fm].q[0] = *(const uint4*)p;
        a[fm].q[1] = *(const uint4*)(p + 8);
      }
      #pragma unroll
      for (int fn = 0; fn < 2; ++fn) {
        const uint32_t* p = &Bs[(wn*32 + fn*16 + ln) * STRIDE + kb + hf*4];
        b[fn].q[0] = *(const uint4*)p;
        b[fn].q[1] = *(const uint4*)(p + 8);
      }
      #pragma unroll
      for (int fm = 0; fm < 4; ++fm)
        #pragma unroll
        for (int fn = 0; fn < 2; ++fn)
          acc[fm][fn] = __builtin_amdgcn_wmma_f32_16x16x32_bf16(
              false, a[fm].v, false, b[fn].v, (short)0, acc[fm][fn], false, false);
    }
  };

  for (int kt = 0; kt < IN_F; kt += 64) {
    __syncthreads();
    { // stage x tile 128x64 f32 -> bf16
      const float4* xr = (const float4*)(x + xbase + kt);
      uint32_t* dst = &As[row * STRIDE + rhf * 16];
      #pragma unroll
      for (int j = 0; j < 4; ++j) {
        float4 f0 = xr[2*j], f1 = xr[2*j+1];
        uint4 o = make_uint4(cvt2(f0.x,f0.y), cvt2(f0.z,f0.w),
                             cvt2(f1.x,f1.y), cvt2(f1.z,f1.w));
        *(uint4*)(dst + 4*j) = o;
      }
    }
    { // stage weight tile 128x64: LDS-LUT NF4 dequant -> bf16 (pk_mul by absmax)
      const int4* qr = (const int4*)(qweight + qbase + (kt >> 1));
      const __bf16 hs = (__bf16)absmax[abase + (kt >> 6)];
      const v2bf sv = {hs, hs};
      uint32_t* dst = &Bs[row * STRIDE + rhf * 16];
      #pragma unroll
      for (int j = 0; j < 4; ++j) {
        int4 q = qr[j];
        uint4 o = make_uint4(pkmul(LUT[q.x & 255], sv), pkmul(LUT[q.y & 255], sv),
                             pkmul(LUT[q.z & 255], sv), pkmul(LUT[q.w & 255], sv));
        *(uint4*)(dst + 4*j) = o;
      }
    }
    if (kt + 64 < IN_F) {
      __builtin_prefetch(x + xbase + kt + 64, 0, 1);
      __builtin_prefetch(qweight + qbase + ((kt + 64) >> 1), 0, 1);
    }
    __syncthreads();
    mma_steps();
  }

  // fused LoRA epilogue: acc += T(128x64) @ lora_B(128x64)^T  (K = RANK = 64)
  __syncthreads();
  {
    const float4* tr = (const float4*)(Tlora + (size_t)(bm*128 + row) * RANK + rhf * 32);
    uint32_t* dst = &As[row * STRIDE + rhf * 16];
    #pragma unroll
    for (int j = 0; j < 4; ++j) {
      float4 f0 = tr[2*j], f1 = tr[2*j+1];
      uint4 o = make_uint4(cvt2(f0.x,f0.y), cvt2(f0.z,f0.w),
                           cvt2(f1.x,f1.y), cvt2(f1.z,f1.w));
      *(uint4*)(dst + 4*j) = o;
    }
    const float4* br = (const float4*)(lora_B + (size_t)(bn*128 + row) * RANK + rhf * 32);
    uint32_t* bdst = &Bs[row * STRIDE + rhf * 16];
    #pragma unroll
    for (int j = 0; j < 4; ++j) {
      float4 f0 = br[2*j], f1 = br[2*j+1];
      uint4 o = make_uint4(cvt2(f0.x,f0.y), cvt2(f0.z,f0.w),
                           cvt2(f1.x,f1.y), cvt2(f1.z,f1.w));
      *(uint4*)(bdst + 4*j) = o;
    }
  }
  __syncthreads();
  mma_steps();

  // store f32 result (C/D layout: lane 0-15 = N, VGPR v = M row, halves +8)
  #pragma unroll
  for (int fn = 0; fn < 2; ++fn) {
    const int ng = bn*128 + wn*32 + fn*16 + ln;
    #pragma unroll
    for (int fm = 0; fm < 4; ++fm)
      #pragma unroll
      for (int v = 0; v < 8; ++v) {
        const int mg = bm*128 + wm*64 + fm*16 + hf*8 + v;
        out[(size_t)mg * OUT_F + ng] = acc[fm][fn][v];
      }
  }
}

extern "C" void kernel_launch(void* const* d_in, const int* in_sizes, int n_in,
                              void* d_out, int out_size, void* d_ws, size_t ws_size,
                              hipStream_t stream) {
  (void)in_sizes; (void)n_in; (void)out_size; (void)ws_size;
  const float* x       = (const float*)d_in[0];
  const int*   qweight = (const int*)d_in[1];
  const float* absmax  = (const float*)d_in[2];
  const float* lora_A  = (const float*)d_in[3];
  const float* lora_B  = (const float*)d_in[4];
  float* out = (float*)d_out;
  float* T   = (float*)d_ws;   // 8192 x 64 f32 = 2 MB scratch

  lora_ax_kernel<<<dim3(M_TOT / 128), 256, 0, stream>>>(x, lora_A, T);
  qgemm_lora_kernel<<<dim3(OUT_F / 128, M_TOT / 128), 256, 0, stream>>>(
      x, qweight, absmax, T, lora_B, out);
}
